// SimplePoseGNN_30751965839401
// MI455X (gfx1250) — matrix operand order repaired and checked by reading, not compiled, not run on hardware
//
#include <hip/hip_runtime.h>
#include <hip/hip_bf16.h>

typedef __attribute__((ext_vector_type(16))) _Float16 v16h;
typedef __attribute__((ext_vector_type(8)))  _Float16 v8h;
typedef __attribute__((ext_vector_type(8)))  float    v8f;

#define BSZ 65536
#define NN  17
#define HD  64
#define KP  1088        // 17*64
#define M1  (BSZ * NN)  // 1,114,112 rows, divisible by 128

// ---- fixed normalized adjacency (from EDGES), as neighbor lists -------------
__constant__ int c_ncnt[NN] = {2,0,0,0,0,4,4,2,2,1,1,3,3,2,2,1,1};
__constant__ int c_nbr[NN][4] = {
  {5,6,0,0},{0,0,0,0},{0,0,0,0},{0,0,0,0},{0,0,0,0},
  {0,6,7,11},{0,5,8,12},{5,9,0,0},{6,10,0,0},{7,0,0,0},{8,0,0,0},
  {5,12,13,0},{6,11,14,0},{11,15,0,0},{12,16,0,0},{13,0,0,0},{14,0,0,0}};

// ---- tiny utility kernels ---------------------------------------------------
__global__ void k_zero(float* p, int n) {
  int i = blockIdx.x * blockDim.x + threadIdx.x;
  if (i < n) p[i] = 0.0f;
}

// Convert weight matrix (K x N, f32 row-major) into WMMA B-fragment order:
// [kt][ntile][lane][e] with 16 contiguous halves per lane -> 2x b128 per read.
__global__ void k_cvt_frag(const float* __restrict__ W, _Float16* __restrict__ Wf,
                           int K, int N) {
  int idx = blockIdx.x * blockDim.x + threadIdx.x;
  int total = (K >> 5) * (N >> 4) * 32;       // one thread per (kt,ntile,lane)
  if (idx >= total) return;
  int lane  = idx & 31;
  int tile  = idx >> 5;                        // kt * (N/16) + ntile
  int nt    = tile % (N >> 4);
  int kt    = tile / (N >> 4);
  int col   = nt * 16 + (lane & 15);
  int kb    = kt * 32 + ((lane < 16) ? 0 : 16);
  _Float16* dst = Wf + (size_t)idx * 16;
  #pragma unroll
  for (int e = 0; e < 16; ++e) dst[e] = (_Float16)W[(size_t)(kb + e) * N + col];
}

// ---- K1: encoder + graph aggregation 1 + BN1 partial stats ------------------
__global__ void k_encode_agg(const float* __restrict__ x, const float* __restrict__ Wenc,
                             const float* __restrict__ benc, _Float16* __restrict__ s1,
                             float* __restrict__ stats /* [0..16]=sum, [17..33]=sumsq */) {
  __shared__ float lsum[NN], lsq[NN];
  int tid = threadIdx.x;
  if (tid < NN) { lsum[tid] = 0.0f; lsq[tid] = 0.0f; }
  __syncthreads();
  long gid = (long)blockIdx.x * blockDim.x + tid;
  float asum = 0.0f, asq = 0.0f;
  int n = 0;
  if (gid < (long)M1) {
    long b = gid / NN;
    n = (int)(gid % NN);
    const float* xb = x + b * (NN * 2);
    float x0 = xb[n * 2], x1 = xb[n * 2 + 1];
    int cnt = c_ncnt[n];
    float w = (cnt > 0) ? 1.0f / (float)cnt : 0.0f;
    float nx0[4], nx1[4];
    #pragma unroll
    for (int j = 0; j < 4; ++j) { int nb = c_nbr[n][j]; nx0[j] = xb[nb*2]; nx1[j] = xb[nb*2+1]; }
    _Float16* srow = s1 + b * KP + n * HD;
    for (int d = 0; d < HD; ++d) {
      float w0 = Wenc[d], w1 = Wenc[HD + d], bb = benc[d];
      float hs = fmaxf(x0 * w0 + x1 * w1 + bb, 0.0f);
      float agg = 0.0f;
      #pragma unroll
      for (int j = 0; j < 4; ++j)
        if (j < cnt) agg += fmaxf(nx0[j] * w0 + nx1[j] * w1 + bb, 0.0f);
      float s = hs + w * agg;
      srow[d] = (_Float16)s;
      asum += s; asq += s * s;
    }
  }
  atomicAdd(&lsum[n], asum);   // ds_add_f32
  atomicAdd(&lsq[n],  asq);
  __syncthreads();
  if (tid < NN) { atomicAdd(&stats[tid], lsum[tid]); atomicAdd(&stats[NN + tid], lsq[tid]); }
}

// ---- K4: graph aggregation 2 + BN2 partial stats ----------------------------
__global__ void k_agg2(const _Float16* __restrict__ g, _Float16* __restrict__ t,
                       float* __restrict__ stats) {
  __shared__ float lsum[NN], lsq[NN];
  int tid = threadIdx.x;
  if (tid < NN) { lsum[tid] = 0.0f; lsq[tid] = 0.0f; }
  __syncthreads();
  long gid = (long)blockIdx.x * blockDim.x + tid;
  float asum = 0.0f, asq = 0.0f;
  int n = 0;
  if (gid < (long)M1) {
    long b = gid / NN;
    n = (int)(gid % NN);
    const _Float16* gb = g + b * KP;
    _Float16* tr = t + b * KP + n * HD;
    int cnt = c_ncnt[n];
    float w = (cnt > 0) ? 1.0f / (float)cnt : 0.0f;
    int nb0 = c_nbr[n][0], nb1 = c_nbr[n][1], nb2 = c_nbr[n][2], nb3 = c_nbr[n][3];
    for (int d = 0; d < HD; ++d) {
      float v = (float)gb[n * HD + d];
      float agg = 0.0f;
      if (cnt > 0) agg += (float)gb[nb0 * HD + d];
      if (cnt > 1) agg += (float)gb[nb1 * HD + d];
      if (cnt > 2) agg += (float)gb[nb2 * HD + d];
      if (cnt > 3) agg += (float)gb[nb3 * HD + d];
      float s = v + w * agg;
      tr[d] = (_Float16)s;
      asum += s; asq += s * s;
    }
  }
  atomicAdd(&lsum[n], asum);
  atomicAdd(&lsq[n],  asq);
  __syncthreads();
  if (tid < NN) { atomicAdd(&stats[tid], lsum[tid]); atomicAdd(&stats[NN + tid], lsq[tid]); }
}

// ---- finalize BN statistics -> per-node affine (scale, shift) ---------------
__global__ void k_finalize(const float* __restrict__ sum, const float* __restrict__ sq,
                           const float* __restrict__ gamma, const float* __restrict__ beta,
                           float* __restrict__ scale, float* __restrict__ shift, float count) {
  int n = threadIdx.x;
  if (n < NN) {
    float mean = sum[n] / count;
    float var  = sq[n] / count - mean * mean;
    float sc   = gamma[n] * rsqrtf(var + 1e-5f);
    scale[n] = sc;
    shift[n] = beta[n] - mean * sc;
  }
}

// ---- WMMA GEMM: O = act(A' x B + bias); A' = optional per-row BN affine -----
// Wave computes a 16x64 output tile; B fragments pre-swizzled (k_cvt_frag),
// so all operand loads are contiguous b128s served from L2. K,N compile-time
// so the k-loop fully unrolls and loads pipeline over the WMMAs.
template <bool BN, bool RELU, bool OUT_F32, int K, int N>
__global__ void k_gemm(const _Float16* __restrict__ A, const _Float16* __restrict__ Bf,
                       const float* __restrict__ bias, const float* __restrict__ scale,
                       const float* __restrict__ shift, _Float16* __restrict__ Oh,
                       float* __restrict__ Of) {
  int lane  = threadIdx.x & 31;
  int wib   = threadIdx.x >> 5;
  int mtile = blockIdx.x * 8 + wib;
  int n0    = blockIdx.y * 64;
  int row0  = mtile * 16;
  int rowA  = row0 + (lane & 15);
  int kbA   = (lane < 16) ? 0 : 16;
  float sc = 1.0f, sh = 0.0f;
  if (BN) { int node = rowA % NN; sc = scale[node]; sh = shift[node]; }
  const _Float16* arow = A + (size_t)rowA * K;
  constexpr int NT = N >> 4;            // total 16-col tiles
  const _Float16* bbase = Bf + ((size_t)(n0 >> 4) * 32 + lane) * 16;

  v8f acc[4];
  #pragma unroll
  for (int nt = 0; nt < 4; ++nt) acc[nt] = (v8f){0.f,0.f,0.f,0.f,0.f,0.f,0.f,0.f};

  constexpr int KSTEPS = K >> 5;
  #pragma unroll
  for (int kt = 0; kt < KSTEPS; ++kt) {
    const int k0 = kt << 5;
    if (KSTEPS > 4 && kt + 2 < KSTEPS)
      __builtin_prefetch(arow + k0 + 64, 0, 3);

    // A fragment: two contiguous 16B runs per lane
    v8h alo = *(const v8h*)(arow + k0 + kbA);
    v8h ahi = *(const v8h*)(arow + k0 + kbA + 8);
    // all four B fragments up front (8x b128, L2-resident weights)
    v16h bt[4];
    #pragma unroll
    for (int nt = 0; nt < 4; ++nt) {
      const _Float16* bp = bbase + ((size_t)kt * NT + nt) * (32 * 16);
      v8h blo = *(const v8h*)bp;
      v8h bhi = *(const v8h*)(bp + 8);
      bt[nt] = __builtin_shufflevector(blo, bhi, 0,1,2,3,4,5,6,7,8,9,10,11,12,13,14,15);
    }
    v16h a;
    if (BN) {
      #pragma unroll
      for (int e = 0; e < 8; ++e) {
        a[e]     = (_Float16)(sc * (float)alo[e] + sh);
        a[e + 8] = (_Float16)(sc * (float)ahi[e] + sh);
      }
    } else {
      a = __builtin_shufflevector(alo, ahi, 0,1,2,3,4,5,6,7,8,9,10,11,12,13,14,15);
    }
    #pragma unroll
    for (int nt = 0; nt < 4; ++nt)
      acc[nt] = __builtin_amdgcn_wmma_f32_16x16x32_f16(
          false, a, false, bt[nt], (short)0, acc[nt], false, false);
  }

  int orow = row0 + ((lane < 16) ? 0 : 8);
  int ocol = n0 + (lane & 15);
  #pragma unroll
  for (int nt = 0; nt < 4; ++nt) {
    int cc = ocol + nt * 16;
    float bb = bias[cc];
    #pragma unroll
    for (int r = 0; r < 8; ++r) {
      float v = acc[nt][r] + bb;
      if (RELU) v = fmaxf(v, 0.0f);
      size_t oidx = (size_t)(orow + r) * N + cc;
      if (OUT_F32) Of[oidx] = v; else Oh[oidx] = (_Float16)v;
    }
  }
}

// ---- per-row L2 normalize (wave per row) ------------------------------------
__global__ void k_norm(const float* __restrict__ emb, float* __restrict__ out) {
  int lane = threadIdx.x & 31;
  int wib  = threadIdx.x >> 5;
  long row = (long)blockIdx.x * 8 + wib;
  const float* er = emb + row * 128;
  float4 v = *(const float4*)(er + lane * 4);
  float ss = v.x * v.x + v.y * v.y + v.z * v.z + v.w * v.w;
  #pragma unroll
  for (int m = 16; m >= 1; m >>= 1) ss += __shfl_xor(ss, m, 32);
  float inv = 1.0f / fmaxf(sqrtf(ss), 1e-12f);
  float4 o = {v.x * inv, v.y * inv, v.z * inv, v.w * inv};
  *(float4*)(out + row * 128 + lane * 4) = o;
}

// -----------------------------------------------------------------------------
extern "C" void kernel_launch(void* const* d_in, const int* in_sizes, int n_in,
                              void* d_out, int out_size, void* d_ws, size_t ws_size,
                              hipStream_t stream) {
  const float* x      = (const float*)d_in[0];
  const float* W_enc  = (const float*)d_in[1];
  const float* b_enc  = (const float*)d_in[2];
  const float* W_gc1  = (const float*)d_in[3];
  const float* b_gc1  = (const float*)d_in[4];
  const float* W_gc2  = (const float*)d_in[5];
  const float* b_gc2  = (const float*)d_in[6];
  const float* gamma1 = (const float*)d_in[7];
  const float* beta1  = (const float*)d_in[8];
  const float* gamma2 = (const float*)d_in[9];
  const float* beta2  = (const float*)d_in[10];
  const float* W_p1   = (const float*)d_in[11];
  const float* b_p1   = (const float*)d_in[12];
  const float* W_p2   = (const float*)d_in[13];
  const float* b_p2   = (const float*)d_in[14];

  char* ws = (char*)d_ws;
  float*     stats = (float*)ws;                          // 136 floats used
  _Float16*  whg1  = (_Float16*)(ws + 1024);              // 64x64   (frag order)
  _Float16*  whg2  = (_Float16*)(ws + 1024 + 8192);       // 64x64   (frag order)
  _Float16*  whp1  = (_Float16*)(ws + 1024 + 16384);      // 1088x256(frag order)
  _Float16*  whp2  = (_Float16*)(ws + 1024 + 16384 + 557056); // 256x128
  size_t bigBytes  = (size_t)BSZ * KP * sizeof(_Float16); // 142,606,336
  char* big0 = ws + 655360;
  char* big1 = big0 + bigBytes;
  _Float16* s1     = (_Float16*)big0;   // stage-1 pre-BN
  _Float16* t2     = (_Float16*)big0;   // reuse after s1 dead
  _Float16* zbuf   = (_Float16*)big0;   // reuse after t2 dead
  _Float16* g1     = (_Float16*)big1;   // gc1 output
  _Float16* pooled = (_Float16*)big1;   // reuse after g1 dead
  float*    emb    = (float*)big1;      // reuse after pooled dead

  float cnt = (float)BSZ * (float)HD;

  // prep: zero stats, convert+swizzle weights to f16 fragment order
  k_zero<<<1, 256, 0, stream>>>(stats, 136);
  k_cvt_frag<<<1,  256, 0, stream>>>(W_gc1, whg1, 64, 64);
  k_cvt_frag<<<1,  256, 0, stream>>>(W_gc2, whg2, 64, 64);
  k_cvt_frag<<<68, 256, 0, stream>>>(W_p1,  whp1, 1088, 256);
  k_cvt_frag<<<8,  256, 0, stream>>>(W_p2,  whp2, 256, 128);

  // stage 1: encoder + agg + stats
  k_encode_agg<<<M1 / 256, 256, 0, stream>>>(x, W_enc, b_enc, s1, stats);
  k_finalize<<<1, 32, 0, stream>>>(stats, stats + 17, gamma1, beta1,
                                   stats + 34, stats + 51, cnt);
  // BN1 + gc1 GEMM (M=B*17, K=64, N=64) via WMMA
  k_gemm<true, true, false, 64, 64><<<dim3(M1 / 128, 1), 256, 0, stream>>>(
      s1, whg1, b_gc1, stats + 34, stats + 51, g1, nullptr);

  // stage 2: agg + stats
  k_agg2<<<M1 / 256, 256, 0, stream>>>(g1, t2, stats + 68);
  k_finalize<<<1, 32, 0, stream>>>(stats + 68, stats + 85, gamma2, beta2,
                                   stats + 102, stats + 119, cnt);
  // BN2 + gc2 GEMM -> pooled (B, 1088) f16
  k_gemm<true, true, false, 64, 64><<<dim3(M1 / 128, 1), 256, 0, stream>>>(
      t2, whg2, b_gc2, stats + 102, stats + 119, pooled, nullptr);

  // pool MLP: (B,1088)x(1088,256) + relu -> z
  k_gemm<false, true, false, 1088, 256><<<dim3(BSZ / 128, 4), 256, 0, stream>>>(
      pooled, whp1, b_p1, nullptr, nullptr, zbuf, nullptr);
  // final: (B,256)x(256,128) -> emb f32
  k_gemm<false, false, true, 256, 128><<<dim3(BSZ / 128, 2), 256, 0, stream>>>(
      zbuf, whp2, b_p2, nullptr, nullptr, nullptr, emb);

  // L2 normalize
  k_norm<<<BSZ / 8, 256, 0, stream>>>(emb, (float*)d_out);
}
